// S6Block_87445534146660
// MI455X (gfx1250) — compile-verified, matrix operand order checked by reading, not run on hardware
//
#include <hip/hip_runtime.h>
#include <hip/hip_bf16.h>

// ---------------------------------------------------------------------------
// S6 / selective-scan block for MI455X (gfx1250, wave32, WMMA).
//   B=4, L=4096, Cin=256, N=512, Cout=256
// GEMMs in bf16 v_wmma_f32_16x16x32_bf16 with fragment-major pre-packed
// weights (one 32B vector load per B fragment); recurrence in fp32 chunked
// scan over interleaved (A,u) float2 pairs (one b64 load per step);
// A_bar computed as sigmoid(-delta) (exact identity, no log1p).
// ---------------------------------------------------------------------------

typedef __attribute__((ext_vector_type(16))) __bf16 v16bf;
typedef __attribute__((ext_vector_type(8)))  float  v8f;

#define CB   4
#define CL   4096
#define CIN  256
#define CN   512
#define COUT 256
#define NALL 2048            // [x_proj | Bm | C | delta] fused width
#define MROWS (CB*CL)        // 16384
#define CHUNK 128
#define NCH  (CL/CHUNK)      // 32

// K-index within a 16x32 bf16 A/B fragment, per documented CDNA5 VGPR layout:
// VGPR i (i<4): K = 2i + 8*kh ; VGPR i (i>=4): K = 16 + 2(i-4) + 8*kh
__device__ __forceinline__ int frag_kbase(int i, int kh) {
    return (i < 4 ? 2 * i : 16 + 2 * (i - 4)) + 8 * kh;
}

// ---------------------------------------------------------------------------
// Kernel 0: convert weights to bf16 AND swizzle into fragment-major layout:
//   WallP[((tile*8 + kstep)*32 + lane)*16 + j]   tile in [0,128), kstep in [0,8)
//   WyP  [((tile*16 + kstep)*32 + lane)*16 + j]  tile in [0,16),  kstep in [0,16)
// so each lane's 16-element B fragment is one contiguous 32-byte block.
// Fused Wall columns: [0,512)=Wx  [512,1024)=Wbc[:, :512] (Bm)
//                     [1024,1536)=Wbc[:, 512:] (C)  [1536,2048)=Wd
// ---------------------------------------------------------------------------
__global__ void pack_weights(const float* __restrict__ Wx,
                             const float* __restrict__ Wbc,
                             const float* __restrict__ Wd,
                             const float* __restrict__ Wy,
                             __bf16* __restrict__ WallP,
                             __bf16* __restrict__ WyP) {
    const int i = blockIdx.x * blockDim.x + threadIdx.x;    // 0 .. 524287
    if (i < CIN * NALL) {
        const int j    = i & 15;
        const int lane = (i >> 4) & 31;
        const int ks   = (i >> 9) & 7;
        const int tile = i >> 12;            // 0..127
        const int kh   = lane >> 4;
        const int k    = ks * 32 + frag_kbase(j >> 1, kh) + (j & 1);
        const int c    = tile * 16 + (lane & 15);
        float v;
        if (c < 512)        v = Wx [k * 512  + c];
        else if (c < 1536)  v = Wbc[k * 1024 + (c - 512)];
        else                v = Wd [k * 512  + (c - 1536)];
        WallP[i] = (__bf16)v;
    }
    if (i < CN * COUT) {
        const int j    = i & 15;
        const int lane = (i >> 4) & 31;
        const int ks   = (i >> 9) & 15;
        const int tile = i >> 13;            // 0..15
        const int kh   = lane >> 4;
        const int k    = ks * 32 + frag_kbase(j >> 1, kh) + (j & 1);
        const int col  = tile * 16 + (lane & 15);
        WyP[i] = (__bf16)Wy[k * COUT + col];
    }
}

// ---------------------------------------------------------------------------
// Kernel 1: fused projection GEMM + elementwise (sigmoid/A_bar/u).
// grid = (MROWS/16, 4), block = 256 (8 waves). Each wave owns one 16x16
// (row, n) footprint and computes the FOUR column-aligned tiles
// (x_proj, Bm, C, delta) so the epilogue is register-local. 32 WMMAs/wave.
// ---------------------------------------------------------------------------
__global__ void proj_kernel(const float*  __restrict__ x,
                            const __bf16* __restrict__ WallP,
                            const float*  __restrict__ bx,
                            const float*  __restrict__ bbc,
                            const float*  __restrict__ bd,
                            float2* __restrict__ AUbuf,
                            float*  __restrict__ Cbuf) {
    const int lane = threadIdx.x & 31;
    const int wave = threadIdx.x >> 5;
    const int mt = blockIdx.x;                 // 0..1023 (M tile)
    const int nt = blockIdx.y * 8 + wave;      // 0..31   (N tile in 512 space)
    const int mrow0 = mt * 16;
    const int mr = lane & 15;                  // A-frag row
    const int kh = lane >> 4;                  // K half selector
    const int col = nt * 16 + (lane & 15);     // this lane's output column (D-frag)

    v8f accX = {}, accB = {}, accC = {}, accD = {};
    const float*  xrow  = x + (size_t)(mrow0 + mr) * CIN;
    // fragment-major weights: tile stride 4096, kstep stride 512, group stride 131072
    const __bf16* wbase = WallP + ((size_t)nt * 8 * 32 + lane) * 16;

#pragma unroll
    for (int ks = 0; ks < 8; ++ks) {
        v16bf a;
#pragma unroll
        for (int i = 0; i < 8; ++i) {
            const int kb = frag_kbase(i, kh);               // even -> 8B aligned
            const float2 xv = *(const float2*)(xrow + ks * 32 + kb);
            a[2*i]   = (__bf16)xv.x;
            a[2*i+1] = (__bf16)xv.y;
        }
        const __bf16* p = wbase + ks * 512;
        const v16bf bX = *(const v16bf*)(p);
        const v16bf bB = *(const v16bf*)(p + 131072);
        const v16bf bC = *(const v16bf*)(p + 262144);
        const v16bf bD = *(const v16bf*)(p + 393216);
        accX = __builtin_amdgcn_wmma_f32_16x16x32_bf16(false, a, false, bX, (short)0, accX, false, false);
        accB = __builtin_amdgcn_wmma_f32_16x16x32_bf16(false, a, false, bB, (short)0, accB, false, false);
        accC = __builtin_amdgcn_wmma_f32_16x16x32_bf16(false, a, false, bC, (short)0, accC, false, false);
        accD = __builtin_amdgcn_wmma_f32_16x16x32_bf16(false, a, false, bD, (short)0, accD, false, false);
    }

    const float bxv = bx[col];
    const float bBv = bbc[col];
    const float bCv = bbc[col + 512];
    const float bdv = bd[col];
#pragma unroll
    for (int r = 0; r < 8; ++r) {
        const int row = mrow0 + r + 8 * kh;      // D-frag: M = r + 8*(lane>=16)
        const float xp = accX[r] + bxv;
        const float bm = accB[r] + bBv;
        const float cv = accC[r] + bCv;
        const float dl = accD[r] + bdv;
        // A_bar = exp(-softplus(d)) == 1/(1+e^d);  1-A_bar == e^d/(1+e^d)
        const float e   = __expf(fminf(dl, 30.0f));   // clamp avoids inf*0
        const float Ab  = __builtin_amdgcn_rcpf(1.0f + e);
        const float u   = (e * Ab) * bm * xp;
        const size_t idx = (size_t)row * CN + col;
        AUbuf[idx] = make_float2(Ab, u);              // one b64 store
        Cbuf[idx]  = cv;
    }
}

// ---------------------------------------------------------------------------
// Kernel 2: per-chunk scan -> carries (prodA, h_end from h0=0).
// One thread per (b, chunk, n): 4*32*512 = 65536 threads.
// One global_load_b64 per timestep (A,u interleaved).
// ---------------------------------------------------------------------------
__global__ void scan_chunks(const float2* __restrict__ AUbuf,
                            float* __restrict__ prodA,
                            float* __restrict__ hend) {
    const int gid = blockIdx.x;                 // 0 .. 255
    const int n = (gid & 1) * 256 + threadIdx.x;
    const int c = (gid >> 1) & (NCH - 1);
    const int b = gid >> 6;
    const size_t base = ((size_t)b * CL + (size_t)c * CHUNK) * CN + n;
    float h = 0.0f, p = 1.0f;
#pragma unroll 4
    for (int t = 0; t < CHUNK; ++t) {
        const float2 au = AUbuf[base + (size_t)t * CN];
        h = fmaf(au.x, h, au.y);
        p *= au.x;
    }
    const size_t ci = ((size_t)b * NCH + c) * CN + n;
    prodA[ci] = p;
    hend[ci]  = h;
}

// ---------------------------------------------------------------------------
// Kernel 3: sequential scan over 32 chunk carries per (b,n) channel.
// hin[b,c,n] = exclusive prefix state entering chunk c.
// ---------------------------------------------------------------------------
__global__ void scan_carries(const float* __restrict__ prodA,
                             const float* __restrict__ hend,
                             float* __restrict__ hin) {
    const int i = blockIdx.x * blockDim.x + threadIdx.x;  // 0..2047
    const int b = i >> 9;
    const int n = i & (CN - 1);
    float h = 0.0f;
    for (int c = 0; c < NCH; ++c) {
        const size_t ci = ((size_t)b * NCH + c) * CN + n;
        hin[ci] = h;
        h = fmaf(prodA[ci], h, hend[ci]);
    }
}

// ---------------------------------------------------------------------------
// Kernel 4: re-scan each chunk from its true incoming state, fuse y = C*h,
// emit bf16 directly as the A-operand of the output GEMM.
// ---------------------------------------------------------------------------
__global__ void scan_apply(const float2* __restrict__ AUbuf,
                           const float*  __restrict__ Cbuf,
                           const float*  __restrict__ hin,
                           __bf16* __restrict__ Yb) {
    const int gid = blockIdx.x;
    const int n = (gid & 1) * 256 + threadIdx.x;
    const int c = (gid >> 1) & (NCH - 1);
    const int b = gid >> 6;
    const size_t base = ((size_t)b * CL + (size_t)c * CHUNK) * CN + n;
    float h = hin[((size_t)b * NCH + c) * CN + n];
#pragma unroll 4
    for (int t = 0; t < CHUNK; ++t) {
        const size_t idx = base + (size_t)t * CN;
        const float2 au = AUbuf[idx];
        h = fmaf(au.x, h, au.y);
        Yb[idx] = (__bf16)(Cbuf[idx] * h);
    }
}

// ---------------------------------------------------------------------------
// Kernel 5: output GEMM  out[16384,256] = Yb[16384,512](bf16) @ WyP + by.
// grid = (1024, 2), block = 256 (8 waves), one 16x16 tile per wave.
// ---------------------------------------------------------------------------
__global__ void out_gemm(const __bf16* __restrict__ Yb,
                         const __bf16* __restrict__ WyP,
                         const float*  __restrict__ by,
                         float* __restrict__ out) {
    const int lane = threadIdx.x & 31;
    const int wave = threadIdx.x >> 5;
    const int mt = blockIdx.x;
    const int nt = blockIdx.y * 8 + wave;       // 0..15
    const int mr = lane & 15;
    const int kh = lane >> 4;
    const __bf16* yrow  = Yb + (size_t)(mt * 16 + mr) * CN;
    const __bf16* wbase = WyP + ((size_t)nt * 16 * 32 + lane) * 16;
    const int col = nt * 16 + (lane & 15);

    v8f acc = {};
#pragma unroll 4
    for (int ks = 0; ks < 16; ++ks) {
        v16bf a;
#pragma unroll
        for (int i = 0; i < 8; ++i) {
            const int kb = frag_kbase(i, kh);   // even -> 4B-aligned bf16 pair
            const __bf16* yp = yrow + ks * 32 + kb;
            a[2*i]   = yp[0];
            a[2*i+1] = yp[1];
        }
        const v16bf bfr = *(const v16bf*)(wbase + ks * 512);
        acc = __builtin_amdgcn_wmma_f32_16x16x32_bf16(false, a, false, bfr, (short)0, acc, false, false);
    }
    const float bias = by[col];
#pragma unroll
    for (int r = 0; r < 8; ++r) {
        const int row = mt * 16 + r + 8 * kh;
        out[(size_t)row * COUT + col] = acc[r] + bias;
    }
}

// ---------------------------------------------------------------------------
// Launch
// ---------------------------------------------------------------------------
extern "C" void kernel_launch(void* const* d_in, const int* in_sizes, int n_in,
                              void* d_out, int out_size, void* d_ws, size_t ws_size,
                              hipStream_t stream) {
    const float* x   = (const float*)d_in[0];
    const float* Wx  = (const float*)d_in[1];
    const float* bx  = (const float*)d_in[2];
    const float* Wbc = (const float*)d_in[3];
    const float* bbc = (const float*)d_in[4];
    const float* Wd  = (const float*)d_in[5];
    const float* bd  = (const float*)d_in[6];
    const float* Wy  = (const float*)d_in[7];
    const float* by  = (const float*)d_in[8];
    float* out = (float*)d_out;

    // workspace layout (bytes)
    char* ws = (char*)d_ws;
    const size_t SZ_WALL  = (size_t)CIN * NALL * 2;          //  1 MB
    const size_t SZ_WY    = (size_t)CN * COUT * 2;           //  256 KB
    const size_t SZ_AU    = (size_t)MROWS * CN * 8;          //  67 MB (A,u pairs)
    const size_t SZ_C     = (size_t)MROWS * CN * 4;          //  33.5 MB
    const size_t SZ_CARRY = (size_t)CB * NCH * CN * 4;       //  256 KB each
    size_t off = 0;
    __bf16* WallP = (__bf16*)(ws + off); off += SZ_WALL;
    __bf16* WyP   = (__bf16*)(ws + off); off += SZ_WY;
    float2* AUbuf = (float2*)(ws + off); off += SZ_AU;
    float*  Cbuf  = (float*) (ws + off); off += SZ_C;
    float*  prodA = (float*) (ws + off); off += SZ_CARRY;
    float*  hend  = (float*) (ws + off); off += SZ_CARRY;
    float*  hin   = (float*) (ws + off); off += SZ_CARRY;
    __bf16* Yb    = (__bf16*)(ws + off); off += (size_t)MROWS * CN * 2;

    // 0) weights -> bf16, fragment-major swizzle
    pack_weights<<<(CIN * NALL + 255) / 256, 256, 0, stream>>>(Wx, Wbc, Wd, Wy, WallP, WyP);

    // 1) fused projection GEMM + elementwise epilogue
    proj_kernel<<<dim3(MROWS / 16, 4), 256, 0, stream>>>(x, WallP, bx, bbc, bd,
                                                         AUbuf, Cbuf);

    // 2) chunked scan: local carries
    scan_chunks<<<CB * NCH * (CN / 256), 256, 0, stream>>>(AUbuf, prodA, hend);

    // 3) carry scan across chunks
    scan_carries<<<(CB * CN) / 256, 256, 0, stream>>>(prodA, hend, hin);

    // 4) apply + y = C*h (bf16)
    scan_apply<<<CB * NCH * (CN / 256), 256, 0, stream>>>(AUbuf, Cbuf, hin, Yb);

    // 5) output GEMM + bias
    out_gemm<<<dim3(MROWS / 16, 2), 256, 0, stream>>>(Yb, WyP, by, out);

    (void)in_sizes; (void)n_in; (void)out_size; (void)ws_size; (void)off;
}
